// QuantLinear_73057393705494
// MI455X (gfx1250) — matrix operand order repaired and checked by reading, not compiled, not run on hardware
//
#include <hip/hip_runtime.h>
#include <stdint.h>

typedef __attribute__((ext_vector_type(16))) _Float16 v16h;
typedef __attribute__((ext_vector_type(2)))  _Float16 v2h;
typedef __attribute__((ext_vector_type(8)))  float    v8f;

#define OUT_DIM 8192
#define IN_DIM  8192
#define BATCH   16
#define GROUP   128
#define KSPLIT  8                            // waves per block, each owns IN/KSPLIT of K
#define KB_PER_WAVE (IN_DIM / KSPLIT / 32)   // 32 WMMA K-steps per wave
#define PF      4                            // software-pipeline prefetch depth

union U32H2 { uint32_t u; v2h h; };
union BFrag { uint32_t u[8]; v16h v; };

// Exact int4 -> fp16 pair. Input t: lo nibble [3:0] = even k, hi nibble [7:4] =
// odd k (reference interleave). (n ^ 8) | 0x6400 == fp16(1024 + (n^8)); a packed
// subtract of 1032 yields the exact signed int4 value.
__device__ __forceinline__ uint32_t nib_pair_to_h2(uint32_t t) {
    uint32_t u = (t & 0xFu) | ((t & 0xF0u) << 12);
    u = (u ^ 0x00080008u) | 0x64006400u;             // -> v_bitop3_b32
    U32H2 c; c.u = u;
    c.h = c.h - (_Float16)1032.0f;                   // v_pk_add_f16, exact
    return c.u;
}

__device__ __forceinline__ v16h unpack_b(uint2 d) {
    BFrag bf;
    #pragma unroll
    for (int jj = 0; jj < 4; ++jj) bf.u[jj]     = nib_pair_to_h2(d.x >> (8 * jj));
    #pragma unroll
    for (int jj = 0; jj < 4; ++jj) bf.u[4 + jj] = nib_pair_to_h2(d.y >> (8 * jj));
    return bf.v;
}

// ---------------------------------------------------------------------------
// Pre-pass: x fp32 -> fp16, swizzled into WMMA A-fragment order in d_ws:
// layout [kb 0..255][lane 0..31][16 halves] so a lane's fragment is one
// contiguous 32B read. A k order used here (k = 16h + s, same as B): for k-pair unit
// u = kk/2: vgpr vv = u&7, lane half h = u>>3.
// ---------------------------------------------------------------------------
__global__ __launch_bounds__(256) void xh_swizzle_kernel(
    const float* __restrict__ x, _Float16* __restrict__ xh)
{
    int pe = blockIdx.x * 256 + threadIdx.x;   // 65536 fp32-pairs total
    int u  = pe & 15;
    int kb = (pe >> 4) & 255;
    int m  = pe >> 12;
    int vv = u & 7;
    int h  = u >> 3;
    float2 xv = *(const float2*)(x + m * IN_DIM + kb * 32 + 2 * u);
    U32H2 pk;
    pk.h = (v2h){ (_Float16)xv.x, (_Float16)xv.y };
    *(uint32_t*)(xh + kb * 512 + (h * 16 + m) * 16 + vv * 2) = pk.u;
}

// ---------------------------------------------------------------------------
// Main: one block per 16-column output tile; 8 waves split K 8 ways, then a
// deterministic LDS reduction combines the 8 fp32 partials.
// ---------------------------------------------------------------------------
__global__ __launch_bounds__(256) void qlin_wmma_kernel(
    const _Float16* __restrict__ xh,    // swizzled fp16 x fragments (d_ws)
    const uint8_t* __restrict__ q,      // packed int4 weights
    const float* __restrict__ scale,    // (64,) per-128-output-rows
    const float* __restrict__ bias,     // (8192,)
    float* __restrict__ out)            // (16, 8192) fp32
{
    __shared__ __align__(16) float red[KSPLIT * 256];   // 8 KB partials

    const int tid   = threadIdx.x;
    const int lane  = tid & 31;
    const int wave  = tid >> 5;        // K-split index
    const int nsub  = lane & 15;       // N within tile
    const int khalf = lane >> 4;       // B-fragment K half

    const int o = blockIdx.x * 16 + nsub;
    const uint8_t* qc = q + (size_t)o * (IN_DIM / 2) + wave * (IN_DIM / 2 / KSPLIT)
                          + khalf * 8;
    const _Float16* av = xh + (wave * KB_PER_WAVE) * 512 + lane * 16;

    v8f acc = {};

    // ---- depth-PF software pipeline: rotating register buffers ----
    v16h a_buf[PF];
    uint2 d_buf[PF];
    #pragma unroll
    for (int i = 0; i < PF; ++i) {
        a_buf[i] = *(const v16h*)(av + i * 512);
        d_buf[i] = *(const uint2*)(qc + i * 16);
    }

    #pragma unroll
    for (int kb = 0; kb < KB_PER_WAVE; ++kb) {
        v16h a = a_buf[kb % PF];
        uint2 d = d_buf[kb % PF];
        if (kb + PF < KB_PER_WAVE) {           // constant-folded by full unroll
            a_buf[kb % PF] = *(const v16h*)(av + (kb + PF) * 512);
            d_buf[kb % PF] = *(const uint2*)(qc + (kb + PF) * 16);
        }
        v16h b = unpack_b(d);
        acc = __builtin_amdgcn_wmma_f32_16x16x32_f16(
            /*neg_a=*/false, a, /*neg_b=*/false, b,
            /*c_mod=*/(short)0, acc, /*reuse_a=*/false, /*reuse_b=*/false);
    }

    // ---- deposit partial tile: red[wave][lane][r] ----
    {
        float4 lo = { acc[0], acc[1], acc[2], acc[3] };
        float4 hi = { acc[4], acc[5], acc[6], acc[7] };
        *(float4*)&red[wave * 256 + lane * 8]     = lo;
        *(float4*)&red[wave * 256 + lane * 8 + 4] = hi;
    }
    __syncthreads();

    // ---- deterministic reduction + epilogue: one output element per thread ----
    const int lane2 = tid & 31;
    const int r2    = tid >> 5;
    float ssum = 0.0f;
    #pragma unroll
    for (int w = 0; w < KSPLIT; ++w)
        ssum += red[w * 256 + lane2 * 8 + r2];

    const int n2 = lane2 & 15;
    const int m2 = (lane2 >> 4) * 8 + r2;     // C layout: vgpr r, hi lanes M=8+r
    const int o2 = blockIdx.x * 16 + n2;
    out[(size_t)m2 * OUT_DIM + o2] = ssum * scale[o2 >> 7] + bias[o2];
}

extern "C" void kernel_launch(void* const* d_in, const int* in_sizes, int n_in,
                              void* d_out, int out_size, void* d_ws, size_t ws_size,
                              hipStream_t stream) {
    const float*   x     = (const float*)d_in[0];
    const uint8_t* q     = (const uint8_t*)d_in[1];
    const float*   scale = (const float*)d_in[2];
    const float*   bias  = (const float*)d_in[3];
    float*         out   = (float*)d_out;
    _Float16*      xh    = (_Float16*)d_ws;     // 256 KB swizzled fp16 copy of x

    xh_swizzle_kernel<<<dim3(256), dim3(256), 0, stream>>>(x, xh);
    qlin_wmma_kernel<<<dim3(OUT_DIM / 16), dim3(256), 0, stream>>>(
        xh, q, scale, bias, out);
}
